// TriangleMultiplicativeModule_23235773071644
// MI455X (gfx1250) — matrix operand order repaired and checked by reading, not compiled, not run on hardware
//
#include <hip/hip_runtime.h>

typedef unsigned int u32;
typedef unsigned short u16;
typedef __attribute__((ext_vector_type(2))) unsigned int u32x2;
typedef __attribute__((ext_vector_type(4))) unsigned int u32x4;
typedef __attribute__((ext_vector_type(4))) float f32x4;
typedef __attribute__((ext_vector_type(16))) __bf16 v16bf;
typedef __attribute__((ext_vector_type(8))) float v8f;

#define DIMC 128

union FragU { v16bf bf; u32 u[8]; };

__device__ __forceinline__ u16 f2bf(float f){
  u32 u = __float_as_uint(f);
  u32 r = u + 0x7fffu + ((u >> 16) & 1u);
  return (u16)(r >> 16);
}
__device__ __forceinline__ u32 pk2(float a, float b){
  return (u32)f2bf(a) | ((u32)f2bf(b) << 16);
}
__device__ __forceinline__ float bfl(u32 h){ return __uint_as_float((h & 0xffffu) << 16); }
__device__ __forceinline__ float bfh(u32 h){ return __uint_as_float(h & 0xffff0000u); }
__device__ __forceinline__ float sigm(float x){ return 1.0f / (1.0f + __expf(-x)); }
__device__ __forceinline__ v8f vzero(){
  v8f z = {0.f,0.f,0.f,0.f,0.f,0.f,0.f,0.f};
  return z;
}

// A fragment (16x32 bf16): lane<16 holds row (lane&15), K={0..15}; lane>=16 K={16..31}.
// LDS layout: row-major, K-pairs packed per u32; strideU = u32s per row; kk2 = Kbase/2.
__device__ __forceinline__ v16bf load_fragA(const u32* A, int strideU, int row, int kk2, int lane){
  FragU f;
  int hi = (lane >> 4) & 1;
  const u32* p = A + row*strideU + kk2 + hi*8;
  #pragma unroll
  for (int v=0;v<4;v++) f.u[v] = p[v];
  #pragma unroll
  for (int v=0;v<4;v++) f.u[4+v] = p[4+v];
  return f.bf;
}
// B fragment (32x16 bf16): lane holds column (lane&15), K = (lane>=16?16:0)+2v..2v+1.
// LDS layout: pair-interleaved [K/2][n] u32s; nStrideU = n count.
__device__ __forceinline__ v16bf load_fragB(const u32* W, int nStrideU, int n, int kk2, int lane){
  FragU f;
  int hi = (lane >> 4) & 1;
  const u32* p = W + (kk2 + hi*8)*nStrideU + n;
  #pragma unroll
  for (int v=0;v<8;v++) f.u[v] = p[v*nStrideU];
  return f.bf;
}
__device__ __forceinline__ v8f wmma_bf(v16bf a, v16bf b, v8f c){
  return __builtin_amdgcn_wmma_f32_16x16x32_bf16(false, a, false, b, (short)0, c, false, false);
}

// stage a 128x128 f32 weight into pair-interleaved bf16 u32s [64 kp][128 n]
__device__ __forceinline__ void stage_weight(u32* Wu, const float* __restrict__ Wg, int t){
  int kp = t >> 2;            // 0..63
  int nh = (t & 3) * 32;      // 0..96
  const f32x4* r0 = (const f32x4*)(Wg + (2*kp)*DIMC + nh);
  const f32x4* r1 = (const f32x4*)(Wg + (2*kp+1)*DIMC + nh);
  #pragma unroll
  for (int p=0;p<8;p++){
    f32x4 a = r0[p], b = r1[p];
    #pragma unroll
    for (int e=0;e<4;e++)
      Wu[kp*DIMC + nh + 4*p + e] = pk2(a[e], b[e]);
  }
}

// one 16-token tile x 4 column tiles, K=128 (4 WMMA each)
__device__ __forceinline__ void gemm4(const u32* Au, const u32* Wu, int ti, int ch, int lane, v8f out[4]){
  int row = ti*16 + (lane & 15);
  #pragma unroll
  for (int c=0;c<4;c++){
    int n = (ch*4 + c)*16 + (lane & 15);
    v8f acc = vzero();
    #pragma unroll
    for (int k=0;k<4;k++){
      v16bf af = load_fragA(Au, 64, row, k*16, lane);
      v16bf bf_ = load_fragB(Wu, DIMC, n, k*16, lane);
      acc = wmma_bf(af, bf_, acc);
    }
    out[c] = acc;
  }
}

// ---------------- Kernel 1: LayerNorm + 5 gated projections ----------------
__global__ __launch_bounds__(256)
void k1_ln_proj(const float* __restrict__ x,
  const float* __restrict__ norm_g, const float* __restrict__ norm_b,
  const float* __restrict__ w_left, const float* __restrict__ b_left,
  const float* __restrict__ w_right, const float* __restrict__ b_right,
  const float* __restrict__ w_lgate, const float* __restrict__ b_lgate,
  const float* __restrict__ w_rgate, const float* __restrict__ b_rgate,
  const float* __restrict__ w_ogate, const float* __restrict__ b_ogate,
  u16* __restrict__ lt, u16* __restrict__ rt, u16* __restrict__ og)
{
  __shared__ u32 xnU[64*64];     // 64 tokens x 128 bf16 (paired)
  __shared__ u32 wU[64*128];     // one weight, pair-interleaved
  __shared__ float gS[128], bS[128], bL[128], bR[128], bLG[128], bRG[128], bOG[128];
  int t = threadIdx.x, lane = t & 31, w = t >> 5;
  int token0 = blockIdx.x * 64;
  int b = token0 >> 16;
  int row1 = (token0 >> 8) & 255;   // first N index (k)
  int col2 = token0 & 255;          // second N index base (multiple of 64)

  if (t < 128){
    gS[t]=norm_g[t]; bS[t]=norm_b[t];
    bL[t]=b_left[t]; bR[t]=b_right[t];
    bLG[t]=b_lgate[t]; bRG[t]=b_rgate[t]; bOG[t]=b_ogate[t];
  }
  // LayerNorm: 4 threads per token
  int tt = t >> 2, q = t & 3, d0 = q*32;
  const f32x4* xp = (const f32x4*)(x + (size_t)(token0 + tt)*DIMC + d0);
  float xv[32]; float s=0.f, ss=0.f;
  #pragma unroll
  for (int p=0;p<8;p++){
    f32x4 v = xp[p];
    #pragma unroll
    for (int e=0;e<4;e++){ float f = v[e]; xv[4*p+e] = f; s += f; ss += f*f; }
  }
  s += __shfl_xor(s,1); ss += __shfl_xor(ss,1);
  s += __shfl_xor(s,2); ss += __shfl_xor(ss,2);
  float mu = s * (1.0f/128.0f);
  float var = ss * (1.0f/128.0f) - mu*mu;
  float rs = rsqrtf(var + 1e-5f);
  __syncthreads();               // params visible
  {
    u32* xr = xnU + tt*64 + q*16;
    #pragma unroll
    for (int p=0;p<16;p++){
      int d = d0 + 2*p;
      float a0 = (xv[2*p]   - mu)*rs*gS[d]   + bS[d];
      float a1 = (xv[2*p+1] - mu)*rs*gS[d+1] + bS[d+1];
      xr[p] = pk2(a0, a1);
    }
  }
  stage_weight(wU, w_left, t);
  __syncthreads();

  int ti = w >> 1, ch = w & 1, hi = lane >> 4;
  v8f accA[4], accB[4];

  // left * sigmoid(lgate) -> lt (transposed [b][d][k][j])
  gemm4(xnU, wU, ti, ch, lane, accA);
  __syncthreads();
  stage_weight(wU, w_lgate, t);
  __syncthreads();
  gemm4(xnU, wU, ti, ch, lane, accB);
  #pragma unroll
  for (int c=0;c<4;c++){
    int n = (ch*4+c)*16 + (lane & 15);
    float bl = bL[n], bg = bLG[n];
    float vals[8];
    #pragma unroll
    for (int r=0;r<8;r++) vals[r] = (accA[c][r] + bl) * sigm(accB[c][r] + bg);
    u32x4 sv;
    sv[0]=pk2(vals[0],vals[1]); sv[1]=pk2(vals[2],vals[3]);
    sv[2]=pk2(vals[4],vals[5]); sv[3]=pk2(vals[6],vals[7]);
    size_t goff = (size_t)(b*DIMC + n)*65536 + (size_t)row1*256 + (size_t)(col2 + ti*16 + hi*8);
    *(u32x4*)(lt + goff) = sv;
  }
  __syncthreads();
  // right * sigmoid(rgate) -> rt (transposed [b][d][k][i])
  stage_weight(wU, w_right, t);
  __syncthreads();
  gemm4(xnU, wU, ti, ch, lane, accA);
  __syncthreads();
  stage_weight(wU, w_rgate, t);
  __syncthreads();
  gemm4(xnU, wU, ti, ch, lane, accB);
  #pragma unroll
  for (int c=0;c<4;c++){
    int n = (ch*4+c)*16 + (lane & 15);
    float bl = bR[n], bg = bRG[n];
    float vals[8];
    #pragma unroll
    for (int r=0;r<8;r++) vals[r] = (accA[c][r] + bl) * sigm(accB[c][r] + bg);
    u32x4 sv;
    sv[0]=pk2(vals[0],vals[1]); sv[1]=pk2(vals[2],vals[3]);
    sv[2]=pk2(vals[4],vals[5]); sv[3]=pk2(vals[6],vals[7]);
    size_t goff = (size_t)(b*DIMC + n)*65536 + (size_t)row1*256 + (size_t)(col2 + ti*16 + hi*8);
    *(u32x4*)(rt + goff) = sv;
  }
  __syncthreads();
  // sigmoid(ogate) -> og (natural [token][d])
  stage_weight(wU, w_ogate, t);
  __syncthreads();
  gemm4(xnU, wU, ti, ch, lane, accA);
  #pragma unroll
  for (int c=0;c<4;c++){
    int n = (ch*4+c)*16 + (lane & 15);
    float bg = bOG[n];
    #pragma unroll
    for (int r=0;r<8;r++){
      int m = ti*16 + r + hi*8;
      og[(size_t)(token0 + m)*DIMC + n] = f2bf(sigm(accA[c][r] + bg));
    }
  }
}

// ---------------- Kernel 2: triangle einsum (batched GEMM per (b,d)) ----------------
// Block: 64(i) x 128(j) per (b,d); double-buffered LDS; 1 barrier per K-step;
// each wave owns one 16-wide j strip -> B fragment reused across 4 WMMAs.
__global__ __launch_bounds__(256)
void k2_einsum(const u16* __restrict__ lt, const u16* __restrict__ rt, u16* __restrict__ mixed)
{
  __shared__ u32 lAU[2][64*16];    // A = R^T tile: [i=64][K pairs=16]
  __shared__ u32 lBU[2][16*128];   // B = L tile: pair-interleaved [K/2=16][j=128]
  int t = threadIdx.x, lane = t & 31, w = t >> 5;
  int wg = blockIdx.x;
  int jb = wg & 1, ib = (wg >> 1) & 3, d = (wg >> 3) & 127, b = wg >> 10;
  size_t baseBD = (size_t)(b*DIMC + d)*65536;
  int ibase = ib*64, jbase = jb*128;
  int kp = t >> 4;          // 0..15 (K pair-row of this thread's staging)
  int e4 = (t & 15) * 4;    // A: i offset
  int e8 = (t & 15) * 8;    // B: j offset
  v8f acc[4] = {vzero(), vzero(), vzero(), vzero()};

  auto stageAB = [&](int buf, int k0){
    const u16* r0 = rt + baseBD + (size_t)(k0 + 2*kp)*256 + ibase + e4;
    u32x2 ra = *(const u32x2*)r0;
    u32x2 rb = *(const u32x2*)(r0 + 256);
    #pragma unroll
    for (int p=0;p<4;p++){
      u32 hra = (p&1) ? (ra[p>>1] >> 16) : (ra[p>>1] & 0xffffu);
      u32 hrb = (p&1) ? (rb[p>>1] >> 16) : (rb[p>>1] & 0xffffu);
      lAU[buf][(e4+p)*16 + kp] = hra | (hrb << 16);         // transpose R -> [i][kpair]
    }
    const u16* l0 = lt + baseBD + (size_t)(k0 + 2*kp)*256 + jbase + e8;
    u32x4 la = *(const u32x4*)l0;
    u32x4 lb = *(const u32x4*)(l0 + 256);
    #pragma unroll
    for (int p=0;p<8;p++){
      u32 hla = (p&1) ? (la[p>>1] >> 16) : (la[p>>1] & 0xffffu);
      u32 hlb = (p&1) ? (lb[p>>1] >> 16) : (lb[p>>1] & 0xffffu);
      lBU[buf][kp*128 + e8 + p] = hla | (hlb << 16);        // pair-interleave L
    }
  };

  stageAB(0, 0);
  __syncthreads();
  #pragma unroll
  for (int s=0; s<8; s++){
    int buf = s & 1;
    if (s < 7) stageAB(buf ^ 1, (s+1)*32);
    v16bf bf_ = load_fragB(lBU[buf], 128, w*16 + (lane & 15), 0, lane);
    #pragma unroll
    for (int ti=0; ti<4; ti++){
      v16bf af = load_fragA(lAU[buf], 16, ti*16 + (lane & 15), 0, lane);
      acc[ti] = wmma_bf(af, bf_, acc[ti]);
    }
    __syncthreads();
  }
  int hi = lane >> 4, n = lane & 15;
  #pragma unroll
  for (int ti=0; ti<4; ti++){
    #pragma unroll
    for (int r=0;r<8;r++){
      mixed[baseBD + (size_t)(ibase + ti*16 + r + hi*8)*256 + jbase + w*16 + n] = f2bf(acc[ti][r]);
    }
  }
}

// ---------------- Kernel 3: output LN * gate, then @ w_out + b_out ----------------
__global__ __launch_bounds__(256)
void k3_out(const u16* __restrict__ mixed, const u16* __restrict__ og,
            const float* __restrict__ onorm_g, const float* __restrict__ onorm_b,
            const float* __restrict__ w_out, const float* __restrict__ b_out,
            float* __restrict__ out)
{
  __shared__ u32 mU[64*64];     // 64 tokens x 128 bf16 (paired)
  __shared__ u32 wU[64*128];
  __shared__ float gS[128], bS[128], boS[128];
  int t = threadIdx.x, lane = t & 31, w = t >> 5;
  int wg = blockIdx.x;
  int j0 = (wg & 3) * 64;
  int i  = (wg >> 2) & 255;
  int b  = wg >> 10;
  size_t tok0 = ((size_t)(b*256 + i))*256 + j0;
  if (t < 128){ gS[t]=onorm_g[t]; bS[t]=onorm_b[t]; boS[t]=b_out[t]; }
  // gather-transpose mixed [d=128][j=64] -> mU [j][d/2]
  {
    int dp = t >> 2;          // 0..63
    int jh = (t & 3) * 16;    // 0,16,32,48
    size_t mrow = (size_t)(b*DIMC + 2*dp)*65536 + (size_t)i*256 + j0 + jh;
    u32 Ar[8], Br[8];
    u32x4 a0 = *(const u32x4*)(mixed + mrow);
    u32x4 a1 = *(const u32x4*)(mixed + mrow + 8);
    u32x4 c0 = *(const u32x4*)(mixed + mrow + 65536);
    u32x4 c1 = *(const u32x4*)(mixed + mrow + 65536 + 8);
    Ar[0]=a0[0];Ar[1]=a0[1];Ar[2]=a0[2];Ar[3]=a0[3];
    Ar[4]=a1[0];Ar[5]=a1[1];Ar[6]=a1[2];Ar[7]=a1[3];
    Br[0]=c0[0];Br[1]=c0[1];Br[2]=c0[2];Br[3]=c0[3];
    Br[4]=c1[0];Br[5]=c1[1];Br[6]=c1[2];Br[7]=c1[3];
    #pragma unroll
    for (int qq=0;qq<16;qq++){
      u32 ha = (qq&1) ? (Ar[qq>>1] >> 16) : (Ar[qq>>1] & 0xffffu);
      u32 hb = (qq&1) ? (Br[qq>>1] >> 16) : (Br[qq>>1] & 0xffffu);
      mU[(jh+qq)*64 + dp] = ha | (hb << 16);
    }
  }
  stage_weight(wU, w_out, t);
  __syncthreads();
  // output LN + out-gate (each 32-channel segment read+written by the same thread)
  {
    int jj = t >> 2, q3 = t & 3, d0 = q3*32;
    u32* mr = mU + jj*64 + (d0 >> 1);
    float xv[32]; float s=0.f, ss=0.f;
    #pragma unroll
    for (int p=0;p<16;p++){
      u32 hv = mr[p];
      float f0 = bfl(hv), f1 = bfh(hv);
      xv[2*p]=f0; xv[2*p+1]=f1; s += f0+f1; ss += f0*f0 + f1*f1;
    }
    s += __shfl_xor(s,1); ss += __shfl_xor(ss,1);
    s += __shfl_xor(s,2); ss += __shfl_xor(ss,2);
    float mu = s * (1.f/128.f);
    float var = ss * (1.f/128.f) - mu*mu;
    float rs = rsqrtf(var + 1e-5f);
    const u32x4* ogp = (const u32x4*)(og + (tok0 + jj)*DIMC + d0);
    u32 ogu[16];
    #pragma unroll
    for (int p=0;p<4;p++){ u32x4 v = ogp[p]; ogu[4*p]=v[0]; ogu[4*p+1]=v[1]; ogu[4*p+2]=v[2]; ogu[4*p+3]=v[3]; }
    #pragma unroll
    for (int p=0;p<16;p++){
      int d = d0 + 2*p;
      float y0 = ((xv[2*p]   - mu)*rs*gS[d]   + bS[d])   * bfl(ogu[p]);
      float y1 = ((xv[2*p+1] - mu)*rs*gS[d+1] + bS[d+1]) * bfh(ogu[p]);
      mr[p] = pk2(y0, y1);
    }
  }
  __syncthreads();
  int ti = w >> 1, ch = w & 1, hi = lane >> 4;
  v8f acc[4];
  gemm4(mU, wU, ti, ch, lane, acc);
  #pragma unroll
  for (int c=0;c<4;c++){
    int n = (ch*4+c)*16 + (lane & 15);
    float bo = boS[n];
    #pragma unroll
    for (int r=0;r<8;r++){
      size_t tok = tok0 + ti*16 + r + hi*8;
      out[tok*DIMC + n] = acc[c][r] + bo;
    }
  }
}

extern "C" void kernel_launch(void* const* d_in, const int* in_sizes, int n_in,
                              void* d_out, int out_size, void* d_ws, size_t ws_size,
                              hipStream_t stream)
{
  (void)in_sizes; (void)n_in; (void)out_size; (void)ws_size;
  const float* x       = (const float*)d_in[0];
  const float* norm_g  = (const float*)d_in[1];
  const float* norm_b  = (const float*)d_in[2];
  const float* w_left  = (const float*)d_in[3];
  const float* b_left  = (const float*)d_in[4];
  const float* w_right = (const float*)d_in[5];
  const float* b_right = (const float*)d_in[6];
  const float* w_lgate = (const float*)d_in[7];
  const float* b_lgate = (const float*)d_in[8];
  const float* w_rgate = (const float*)d_in[9];
  const float* b_rgate = (const float*)d_in[10];
  const float* w_ogate = (const float*)d_in[11];
  const float* b_ogate = (const float*)d_in[12];
  const float* onorm_g = (const float*)d_in[13];
  const float* onorm_b = (const float*)d_in[14];
  const float* w_out   = (const float*)d_in[15];
  const float* b_out   = (const float*)d_in[16];

  const size_t HALF = 33554432ull;  // 4*128*256*256 bf16 elems per intermediate
  u16* lt = (u16*)d_ws;             // left  [b][d][k][j]
  u16* rt = lt + HALF;              // right [b][d][k][i]
  u16* og = rt + HALF;              // out-gate [token][d]
  u16* mx = og + HALF;              // mixed [b][d][i][j]

  k1_ln_proj<<<4096, 256, 0, stream>>>(x, norm_g, norm_b, w_left, b_left, w_right, b_right,
      w_lgate, b_lgate, w_rgate, b_rgate, w_ogate, b_ogate, lt, rt, og);
  k2_einsum<<<4096, 256, 0, stream>>>(lt, rt, mx);
  k3_out<<<4096, 256, 0, stream>>>(mx, og, onorm_g, onorm_b, w_out, b_out, (float*)d_out);
}
